// STTConformerBlock_80994493268430
// MI455X (gfx1250) — matrix operand order repaired and checked by reading, not compiled, not run on hardware
//
#include <hip/hip_runtime.h>
#include <cstdint>

// ---------------------------------------------------------------------------
// STT Conformer block for MI455X (gfx1250, wave32, WMMA).
// All heavy GEMMs run on v_wmma_f32_16x16x32_bf16 (f32 accumulate).
// Attention is flash-style (no materialized TxT scores).
// A/K tiles staged with gfx1250 async global->LDS copies when available.
// ---------------------------------------------------------------------------

typedef __attribute__((ext_vector_type(16))) __bf16 v16bf;
typedef __attribute__((ext_vector_type(8)))  float  v8f;
typedef __attribute__((ext_vector_type(4)))  int    v4i;

#define BT_ROWS 8192   // B*T
#define DMODEL  512
#define NHEAD   8
#define TSEQ    2048
#define BATCH   4
#define DHEAD   64
#define FFINNER 2048
#define KCONV   31
#define LN_EPS  1e-5f

#if defined(__has_builtin)
#  if __has_builtin(__builtin_amdgcn_global_load_async_to_lds_b128) && \
      __has_builtin(__builtin_amdgcn_s_wait_asynccnt)
#    define HAVE_ASYNC_LDS 1
#  endif
#endif
#ifndef HAVE_ASYNC_LDS
#  define HAVE_ASYNC_LDS 0
#endif

#if HAVE_ASYNC_LDS
typedef __attribute__((address_space(1))) v4i* as1_v4i_ptr;
typedef __attribute__((address_space(3))) v4i* as3_v4i_ptr;
#endif

static __device__ __forceinline__ float siluf(float v) { return v / (1.f + __expf(-v)); }
static __device__ __forceinline__ float sigmf(float v) { return 1.f / (1.f + __expf(-v)); }

// 16-byte global -> LDS stage: async (ASYNCcnt) when the builtin exists.
static __device__ __forceinline__ void stage16(const __bf16* g, __bf16* l) {
#if HAVE_ASYNC_LDS
    __builtin_amdgcn_global_load_async_to_lds_b128((as1_v4i_ptr)g, (as3_v4i_ptr)l, 0, 0);
#else
    *(uint4*)l = *(const uint4*)g;
#endif
}
static __device__ __forceinline__ void stage_wait() {
#if HAVE_ASYNC_LDS
    __builtin_amdgcn_s_wait_asynccnt(0);
#endif
}

// ---- WMMA fragment loaders (CDNA5 ISA 7.12.2 wave32 layouts) ---------------
// A (16x32 bf16, row-major storage, leading dim = ld):
//   lanes 0-15 : row = lane,    elems 0-7 = K0-7,  elems 8-15 = K16-23
//   lanes 16-31: row = lane-16, elems 0-7 = K8-15, elems 8-15 = K24-31
static __device__ __forceinline__ v16bf frag_a_rowmajor(const __bf16* base, int ld) {
    int l = threadIdx.x & 31;
    const __bf16* r = base + (l & 15) * ld + ((l & 16) ? 8 : 0);
    v16bf f;
#pragma unroll
    for (int j = 0; j < 8; ++j) { f[j] = r[j]; f[8 + j] = r[16 + j]; }
    return f;
}
// B (32x16 bf16) from column-major storage Bt[n][k] (leading dim = ld):
//   lanes 0-15 : col = lane,    elems 0-7 = K0-7,  elems 8-15 = K16-23
//   lanes 16-31: col = lane-16, elems 0-7 = K8-15, elems 8-15 = K24-31
static __device__ __forceinline__ v16bf frag_b_colmajor(const __bf16* base, int ld) {
    int l = threadIdx.x & 31;
    const __bf16* r = base + (l & 15) * ld + ((l & 16) ? 8 : 0);
    v16bf f;
#pragma unroll
    for (int j = 0; j < 8; ++j) { f[j] = r[j]; f[8 + j] = r[16 + j]; }
    return f;
}

// ---------------------------------------------------------------------------
// Generic tiled WMMA GEMM: C[M,N] = A[M,K](bf16) x B[K,N](bf16) + bias
// Block: 256 threads (8 waves). Tile: 128x128, K-step 64 (32 KB LDS).
// Wave w computes rows [w*16, w*16+16) x all 128 columns (8 wmma accumulators)
// EPI: 0 = f32 out (+bias)        1 = bf16 out = silu(acc+bias)
//      2 = resid += alpha*(acc+b) 3 = bf16 out (+bias)
// ---------------------------------------------------------------------------
template <int EPI>
__global__ __launch_bounds__(256) void gemm_bf16(
    const __bf16* __restrict__ A, const __bf16* __restrict__ Bm,
    const float* __restrict__ bias, float alpha,
    float* __restrict__ outf, __bf16* __restrict__ outb,
    float* __restrict__ resid, int M, int N, int K) {
    __shared__ __align__(16) __bf16 As[128 * 64];   // row-major [m][k]
    __shared__ __align__(16) __bf16 Bs[128 * 64];   // transposed [n][k]

    const int tid  = threadIdx.x;
    const int wave = tid >> 5;
    const int m0   = blockIdx.y * 128;
    const int n0   = blockIdx.x * 128;

    v8f acc[8];
    const v8f zero = {0.f, 0.f, 0.f, 0.f, 0.f, 0.f, 0.f, 0.f};
#pragma unroll
    for (int nt = 0; nt < 8; ++nt) acc[nt] = zero;

    for (int k0 = 0; k0 < K; k0 += 64) {
        __syncthreads();
        // A tile: 128x64 = 1024 x 16B chunks, async global->LDS when available.
#pragma unroll
        for (int c = tid; c < 1024; c += 256) {
            const int r = c >> 3, seg = c & 7;
            stage16(&A[(size_t)(m0 + r) * K + k0 + seg * 8], &As[r * 64 + seg * 8]);
        }
        // B tile: 64x128 read coalesced two K-rows at a time; K-pairs packed
        // into dwords so the transposed LDS store is ds_store_b32.
#pragma unroll
        for (int c = tid; c < 512; c += 256) {
            const int r2 = c >> 4, seg = c & 15;  // k = 2*r2, cols seg*8..seg*8+7
            uint4 u0 = *(const uint4*)&Bm[(size_t)(k0 + 2 * r2) * N + n0 + seg * 8];
            uint4 u1 = *(const uint4*)&Bm[(size_t)(k0 + 2 * r2 + 1) * N + n0 + seg * 8];
            const __bf16* e0 = (const __bf16*)&u0;
            const __bf16* e1 = (const __bf16*)&u1;
#pragma unroll
            for (int j = 0; j < 8; ++j) {
                union { __bf16 h[2]; unsigned u; } p;
                p.h[0] = e0[j];
                p.h[1] = e1[j];
                *(unsigned*)&Bs[(seg * 8 + j) * 64 + 2 * r2] = p.u;
            }
        }
        if (k0 + 64 < K) {  // gfx1250 global_prefetch_b8 for the next tiles
            __builtin_prefetch(&A[(size_t)(m0 + (tid >> 1)) * K + k0 + 64], 0, 1);
            __builtin_prefetch(&Bm[(size_t)(k0 + 64 + (tid >> 2)) * N + n0], 0, 1);
        }
        stage_wait();
        __syncthreads();

        const __bf16* arow = &As[(wave * 16) * 64];
#pragma unroll
        for (int s = 0; s < 2; ++s) {
            v16bf a = frag_a_rowmajor(arow + s * 32, 64);
#pragma unroll
            for (int nt = 0; nt < 8; ++nt) {
                v16bf b = frag_b_colmajor(&Bs[(nt * 16) * 64 + s * 32], 64);
                acc[nt] = __builtin_amdgcn_wmma_f32_16x16x32_bf16(
                    false, a, false, b, (short)0, acc[nt], false, false);
            }
        }
    }

    // Epilogue. C layout: VGPR r -> (M = mb + r, N = lane&15), mb = (lane<16)?0:8
    const int lane = tid & 31;
    const int ncol = lane & 15;
    const int mb   = (lane & 16) ? 8 : 0;
    const int mrow = m0 + wave * 16 + mb;
#pragma unroll
    for (int nt = 0; nt < 8; ++nt) {
        const int n  = n0 + nt * 16 + ncol;
        const float bv = bias ? bias[n] : 0.f;
#pragma unroll
        for (int r = 0; r < 8; ++r) {
            const size_t idx = (size_t)(mrow + r) * N + n;
            const float v = acc[nt][r] + bv;
            if (EPI == 0)      outf[idx] = v;
            else if (EPI == 1) outb[idx] = (__bf16)siluf(v);
            else if (EPI == 2) resid[idx] += alpha * v;
            else               outb[idx] = (__bf16)v;
        }
    }
}

// ---------------------------------------------------------------------------
// LayerNorm over D=512 per row; optional bf16 and/or f32 outputs.
// ---------------------------------------------------------------------------
__global__ __launch_bounds__(256) void layernorm_k(
    const float* __restrict__ x, const float* __restrict__ g,
    const float* __restrict__ b, __bf16* __restrict__ obf,
    float* __restrict__ of32) {
    const int row = blockIdx.x, tid = threadIdx.x;
    const float* xr = x + (size_t)row * DMODEL;
    float v0 = xr[tid], v1 = xr[tid + 256];
    __shared__ float ssum[256], ssq[256];
    ssum[tid] = v0 + v1;
    ssq[tid]  = v0 * v0 + v1 * v1;
    __syncthreads();
    for (int o = 128; o > 0; o >>= 1) {
        if (tid < o) { ssum[tid] += ssum[tid + o]; ssq[tid] += ssq[tid + o]; }
        __syncthreads();
    }
    const float mean = ssum[0] * (1.f / DMODEL);
    const float var  = ssq[0] * (1.f / DMODEL) - mean * mean;
    const float rstd = rsqrtf(var + LN_EPS);
    const float y0 = (v0 - mean) * rstd * g[tid] + b[tid];
    const float y1 = (v1 - mean) * rstd * g[tid + 256] + b[tid + 256];
    if (obf)  { obf[(size_t)row * DMODEL + tid] = (__bf16)y0;
                obf[(size_t)row * DMODEL + tid + 256] = (__bf16)y1; }
    if (of32) { of32[(size_t)row * DMODEL + tid] = y0;
                of32[(size_t)row * DMODEL + tid + 256] = y1; }
}

// ---------------------------------------------------------------------------
// RoPE (optional) + transpose [B*T, H*DH](f32) -> [B,H,T,DH](bf16).
// One block per (b,t); thread = (head, pair).
// ---------------------------------------------------------------------------
__global__ __launch_bounds__(256) void rope_transpose_k(
    const float* __restrict__ in, __bf16* __restrict__ outp, int do_rope) {
    const int bt = blockIdx.x;
    const int b = bt / TSEQ, t = bt % TSEQ;
    const int h = threadIdx.x >> 5, i = threadIdx.x & 31;
    const float* src = in + (size_t)bt * DMODEL + h * DHEAD;
    float xe = src[2 * i], xo = src[2 * i + 1], o0, o1;
    if (do_rope) {
        const float freq = __powf(10000.f, -(float)(2 * i) * (1.f / DHEAD));
        const float ang = (float)t * freq;
        const float c = __cosf(ang), s = __sinf(ang);
        o0 = xe * c - xo * s;
        o1 = xe * s + xo * c;
    } else { o0 = xe; o1 = xo; }
    __bf16* dst = outp + (((size_t)(b * NHEAD + h) * TSEQ + t) * DHEAD);
    dst[2 * i] = (__bf16)o0;
    dst[2 * i + 1] = (__bf16)o1;
}

// ---------------------------------------------------------------------------
// Flash attention, wave32 WMMA. Grid: (B*H, T/128). Block: 256 (8 waves).
// Wave w owns 16 query rows; loops over 32-key tiles with online softmax.
// Q/K/V layout [B,H,T,DH] bf16; output written as [B,T,H*DH] bf16.
// ---------------------------------------------------------------------------
__global__ __launch_bounds__(256) void flash_attn_k(
    const __bf16* __restrict__ Q, const __bf16* __restrict__ Kt,
    const __bf16* __restrict__ Vt, __bf16* __restrict__ O) {
    __shared__ __align__(16) __bf16 Ks[32 * 64];     // [key][d]
    __shared__ __align__(16) __bf16 Vs2[64 * 32];    // transposed [d][key]
    __shared__ __align__(16) __bf16 Ps[8][16 * 32];  // per-wave P scratch

    const int bh = blockIdx.x;
    const int b = bh / NHEAD, h = bh % NHEAD;
    const int tid = threadIdx.x, wave = tid >> 5, lane = tid & 31;
    const int q0 = blockIdx.y * 128 + wave * 16;
    const float sc = 0.125f;  // 1/sqrt(DH)

    const __bf16* Qbase = Q + ((size_t)bh * TSEQ + q0) * DHEAD;
    const v16bf qa0 = frag_a_rowmajor(Qbase, DHEAD);
    const v16bf qa1 = frag_a_rowmajor(Qbase + 32, DHEAD);

    const v8f zero = {0.f, 0.f, 0.f, 0.f, 0.f, 0.f, 0.f, 0.f};
    v8f o[4];
    float m[8], l[8];
#pragma unroll
    for (int nt = 0; nt < 4; ++nt) o[nt] = zero;
#pragma unroll
    for (int r = 0; r < 8; ++r) { m[r] = -3.0e38f; l[r] = 0.f; }

    const int ncol = lane & 15;

    for (int kt = 0; kt < TSEQ / 32; ++kt) {
        __syncthreads();
        {   // stage K row-major (async path) and V transposed (packed pairs)
            const __bf16* Kg = Kt + ((size_t)bh * TSEQ + kt * 32) * DHEAD;
            const __bf16* Vg = Vt + ((size_t)bh * TSEQ + kt * 32) * DHEAD;
            const int r = tid >> 3, seg = tid & 7;
            stage16(&Kg[(size_t)r * 64 + seg * 8], &Ks[r * 64 + seg * 8]);
            if (tid < 128) {
                const int r2 = tid >> 3, sg = tid & 7;  // keys 2*r2..2*r2+1
                uint4 u0 = *(const uint4*)&Vg[(size_t)(2 * r2) * 64 + sg * 8];
                uint4 u1 = *(const uint4*)&Vg[(size_t)(2 * r2 + 1) * 64 + sg * 8];
                const __bf16* e0 = (const __bf16*)&u0;
                const __bf16* e1 = (const __bf16*)&u1;
#pragma unroll
                for (int j = 0; j < 8; ++j) {
                    union { __bf16 hh[2]; unsigned u; } p;
                    p.hh[0] = e0[j];
                    p.hh[1] = e1[j];
                    *(unsigned*)&Vs2[(sg * 8 + j) * 32 + 2 * r2] = p.u;
                }
            }
        }
        stage_wait();
        __syncthreads();

        // S = Q . K^T : two 16x16 key halves, K-dim 64 = 2 wmma steps
        v8f s0 = zero, s1 = zero;
#pragma unroll
        for (int s = 0; s < 2; ++s) {
            const v16bf qa = (s == 0) ? qa0 : qa1;
            v16bf bk0 = frag_b_colmajor(&Ks[0 * 64 + s * 32], 64);
            v16bf bk1 = frag_b_colmajor(&Ks[16 * 64 + s * 32], 64);
            s0 = __builtin_amdgcn_wmma_f32_16x16x32_bf16(false, qa, false, bk0, (short)0, s0, false, false);
            s1 = __builtin_amdgcn_wmma_f32_16x16x32_bf16(false, qa, false, bk1, (short)0, s1, false, false);
        }

        // online softmax (rows live across one 16-lane half)
#pragma unroll
        for (int r = 0; r < 8; ++r) {
            float a0 = s0[r] * sc, a1 = s1[r] * sc;
            float rowm = fmaxf(a0, a1);
#pragma unroll
            for (int off = 1; off < 16; off <<= 1)
                rowm = fmaxf(rowm, __shfl_xor(rowm, off, 32));
            const float mnew = fmaxf(m[r], rowm);
            const float corr = __expf(m[r] - mnew);
            const float p0 = __expf(a0 - mnew), p1 = __expf(a1 - mnew);
            float rs = p0 + p1;
#pragma unroll
            for (int off = 1; off < 16; off <<= 1)
                rs += __shfl_xor(rs, off, 32);
            l[r] = l[r] * corr + rs;
            m[r] = mnew;
#pragma unroll
            for (int nt = 0; nt < 4; ++nt) o[nt][r] *= corr;
            const int prow = ((lane & 16) ? 8 : 0) + r;
            Ps[wave][prow * 32 + ncol]      = (__bf16)p0;
            Ps[wave][prow * 32 + 16 + ncol] = (__bf16)p1;
        }
        __syncthreads();

        // O += P(16x32) . V(32x64)
        v16bf pa = frag_a_rowmajor(&Ps[wave][0], 32);
#pragma unroll
        for (int nt = 0; nt < 4; ++nt) {
            v16bf vb = frag_b_colmajor(&Vs2[(nt * 16) * 32], 32);
            o[nt] = __builtin_amdgcn_wmma_f32_16x16x32_bf16(false, pa, false, vb, (short)0, o[nt], false, false);
        }
    }

    // normalize and store as [B, T, H*DH]
    const int mb = (lane & 16) ? 8 : 0;
#pragma unroll
    for (int nt = 0; nt < 4; ++nt) {
#pragma unroll
        for (int r = 0; r < 8; ++r) {
            const float v = o[nt][r] / l[r];
            const size_t t = (size_t)(q0 + mb + r);
            O[((size_t)b * TSEQ + t) * DMODEL + h * DHEAD + nt * 16 + ncol] = (__bf16)v;
        }
    }
}

// ---------------------------------------------------------------------------
// Small elementwise / reduction kernels
// ---------------------------------------------------------------------------
__global__ void cvt_f32_bf16_k(const float* __restrict__ in, __bf16* __restrict__ outp, size_t n) {
    for (size_t i = (size_t)blockIdx.x * blockDim.x + threadIdx.x; i < n;
         i += (size_t)gridDim.x * blockDim.x)
        outp[i] = (__bf16)in[i];
}

__global__ void glu_k(const float* __restrict__ u, float* __restrict__ outp, size_t n) {
    for (size_t i = (size_t)blockIdx.x * blockDim.x + threadIdx.x; i < n;
         i += (size_t)gridDim.x * blockDim.x) {
        const size_t row = i >> 9;
        const int c = (int)(i & 511);
        const float a = u[row * 1024 + c];
        const float g = u[row * 1024 + 512 + c];
        outp[i] = a * sigmf(g);
    }
}

__global__ __launch_bounds__(256) void dwconv_k(
    const float* __restrict__ hc, const float* __restrict__ w,
    const float* __restrict__ db, float* __restrict__ g) {
    const int bt = blockIdx.x;
    const int b = bt / TSEQ, t = bt % TSEQ;
#pragma unroll
    for (int e = 0; e < 2; ++e) {
        const int c = threadIdx.x + e * 256;
        float acc = db[c];
#pragma unroll
        for (int k = 0; k < KCONV; ++k) {
            const int tt = t + k - KCONV / 2;
            if (tt >= 0 && tt < TSEQ)
                acc += hc[((size_t)b * TSEQ + tt) * DMODEL + c] * w[c * KCONV + k];
        }
        g[(size_t)bt * DMODEL + c] = acc;
    }
}

__global__ __launch_bounds__(256) void bn_stats_k(
    const float* __restrict__ g, float* __restrict__ mean, float* __restrict__ inv) {
    const int c = blockIdx.x;
    float s = 0.f, sq = 0.f;
    for (int i = threadIdx.x; i < BT_ROWS; i += 256) {
        const float v = g[(size_t)i * DMODEL + c];
        s += v; sq += v * v;
    }
    __shared__ float a[256], b2[256];
    a[threadIdx.x] = s; b2[threadIdx.x] = sq;
    __syncthreads();
    for (int o = 128; o > 0; o >>= 1) {
        if (threadIdx.x < o) { a[threadIdx.x] += a[threadIdx.x + o]; b2[threadIdx.x] += b2[threadIdx.x + o]; }
        __syncthreads();
    }
    if (threadIdx.x == 0) {
        const float mu = a[0] * (1.f / BT_ROWS);
        const float var = b2[0] * (1.f / BT_ROWS) - mu * mu;
        mean[c] = mu;
        inv[c]  = rsqrtf(var + LN_EPS);
    }
}

__global__ void bn_apply_silu_k(
    const float* __restrict__ g, const float* __restrict__ mean,
    const float* __restrict__ inv, const float* __restrict__ gam,
    const float* __restrict__ bet, __bf16* __restrict__ outp, size_t n) {
    for (size_t i = (size_t)blockIdx.x * blockDim.x + threadIdx.x; i < n;
         i += (size_t)gridDim.x * blockDim.x) {
        const int c = (int)(i & 511);
        const float v = (g[i] - mean[c]) * inv[c] * gam[c] + bet[c];
        outp[i] = (__bf16)siluf(v);
    }
}

// ---------------------------------------------------------------------------
// Launch sequence
// ---------------------------------------------------------------------------
extern "C" void kernel_launch(void* const* d_in, const int* in_sizes, int n_in,
                              void* d_out, int out_size, void* d_ws, size_t ws_size,
                              hipStream_t stream) {
    (void)in_sizes; (void)n_in; (void)out_size; (void)ws_size;
    const float* x        = (const float*)d_in[0];
    const float* ff1_lng  = (const float*)d_in[1];
    const float* ff1_lnb  = (const float*)d_in[2];
    const float* ff1_w1   = (const float*)d_in[3];
    const float* ff1_b1   = (const float*)d_in[4];
    const float* ff1_w2   = (const float*)d_in[5];
    const float* ff1_b2   = (const float*)d_in[6];
    const float* attn_lng = (const float*)d_in[7];
    const float* attn_lnb = (const float*)d_in[8];
    const float* wq       = (const float*)d_in[9];
    const float* wk       = (const float*)d_in[10];
    const float* wv       = (const float*)d_in[11];
    const float* wo       = (const float*)d_in[12];
    const float* conv_lng = (const float*)d_in[13];
    const float* conv_lnb = (const float*)d_in[14];
    const float* pw1_w    = (const float*)d_in[15];
    const float* pw1_b    = (const float*)d_in[16];
    const float* dw_w     = (const float*)d_in[17];
    const float* dw_b     = (const float*)d_in[18];
    const float* bn_g     = (const float*)d_in[19];
    const float* bn_b     = (const float*)d_in[20];
    const float* pw2_w    = (const float*)d_in[21];
    const float* pw2_b    = (const float*)d_in[22];
    const float* ff2_lng  = (const float*)d_in[23];
    const float* ff2_lnb  = (const float*)d_in[24];
    const float* ff2_w1   = (const float*)d_in[25];
    const float* ff2_b1   = (const float*)d_in[26];
    const float* ff2_w2   = (const float*)d_in[27];
    const float* ff2_b2   = (const float*)d_in[28];
    const float* out_lng  = (const float*)d_in[29];
    const float* out_lnb  = (const float*)d_in[30];
    float* out = (float*)d_out;

    // --- workspace layout ---
    char* base = (char*)d_ws;
    size_t off = 0;
    auto take = [&](size_t bytes) -> void* {
        void* p = base + off;
        off += (bytes + 255) & ~(size_t)255;
        return p;
    };
    const size_t M = BT_ROWS;
    float*  xw      = (float*)take(M * DMODEL * 4);           // f32 residual stream
    __bf16* lnbf    = (__bf16*)take(M * DMODEL * 2);          // LN output (bf16 GEMM A)
    __bf16* wb_ff11 = (__bf16*)take((size_t)DMODEL * FFINNER * 2);
    __bf16* wb_ff12 = (__bf16*)take((size_t)FFINNER * DMODEL * 2);
    __bf16* wb_q    = (__bf16*)take((size_t)DMODEL * DMODEL * 2);
    __bf16* wb_k    = (__bf16*)take((size_t)DMODEL * DMODEL * 2);
    __bf16* wb_v    = (__bf16*)take((size_t)DMODEL * DMODEL * 2);
    __bf16* wb_o    = (__bf16*)take((size_t)DMODEL * DMODEL * 2);
    __bf16* wb_pw1  = (__bf16*)take((size_t)DMODEL * 2 * DMODEL * 2);
    __bf16* wb_pw2  = (__bf16*)take((size_t)DMODEL * DMODEL * 2);
    __bf16* wb_ff21 = (__bf16*)take((size_t)DMODEL * FFINNER * 2);
    __bf16* wb_ff22 = (__bf16*)take((size_t)FFINNER * DMODEL * 2);
    float*  bn_mean = (float*)take(DMODEL * 4);
    float*  bn_inv  = (float*)take(DMODEL * 4);
    char* scratch = (char*)take(80ull << 20);                 // phase-shared scratch

    __bf16* inner_bf = (__bf16*)scratch;                       // [M, 2048] bf16 (32 MiB)
    float*  qf       = (float*)scratch;                        // [M, 512] f32  (16 MiB)
    __bf16* qb       = (__bf16*)(scratch + (16ull << 20));     // [B,H,T,64] bf16 (8 MiB)
    __bf16* kb       = (__bf16*)(scratch + (24ull << 20));
    __bf16* vb       = (__bf16*)(scratch + (32ull << 20));
    __bf16* attnbf   = (__bf16*)(scratch + (40ull << 20));     // [M, 512] bf16
    float*  ubuf     = (float*)scratch;                        // [M, 1024] f32 (32 MiB)
    float*  hcbuf    = (float*)(scratch + (32ull << 20));      // [M, 512] f32
    float*  gbuf     = (float*)(scratch + (48ull << 20));      // [M, 512] f32
    __bf16* sbf      = (__bf16*)(scratch + (64ull << 20));     // [M, 512] bf16

    // --- one-time (per launch) weight conversion to bf16 ---
    auto cvt = [&](const float* src, __bf16* dst, size_t n) {
        cvt_f32_bf16_k<<<1024, 256, 0, stream>>>(src, dst, n);
    };
    cvt(ff1_w1, wb_ff11, (size_t)DMODEL * FFINNER);
    cvt(ff1_w2, wb_ff12, (size_t)FFINNER * DMODEL);
    cvt(wq, wb_q, (size_t)DMODEL * DMODEL);
    cvt(wk, wb_k, (size_t)DMODEL * DMODEL);
    cvt(wv, wb_v, (size_t)DMODEL * DMODEL);
    cvt(wo, wb_o, (size_t)DMODEL * DMODEL);
    cvt(pw1_w, wb_pw1, (size_t)DMODEL * 2 * DMODEL);
    cvt(pw2_w, wb_pw2, (size_t)DMODEL * DMODEL);
    cvt(ff2_w1, wb_ff21, (size_t)DMODEL * FFINNER);
    cvt(ff2_w2, wb_ff22, (size_t)FFINNER * DMODEL);

    (void)hipMemcpyAsync(xw, x, M * DMODEL * sizeof(float), hipMemcpyDeviceToDevice, stream);

    const dim3 blk(256);
    const dim3 gD(DMODEL / 128, M / 128);     // N=512
    const dim3 gI(FFINNER / 128, M / 128);    // N=2048
    const dim3 gP(2 * DMODEL / 128, M / 128); // N=1024

    // === FF1 (half-step) ===
    layernorm_k<<<M, blk, 0, stream>>>(xw, ff1_lng, ff1_lnb, lnbf, nullptr);
    gemm_bf16<1><<<gI, blk, 0, stream>>>(lnbf, wb_ff11, ff1_b1, 0.f, nullptr, inner_bf, nullptr, M, FFINNER, DMODEL);
    gemm_bf16<2><<<gD, blk, 0, stream>>>(inner_bf, wb_ff12, ff1_b2, 0.5f, nullptr, nullptr, xw, M, DMODEL, FFINNER);

    // === Multi-head self-attention with RoPE ===
    layernorm_k<<<M, blk, 0, stream>>>(xw, attn_lng, attn_lnb, lnbf, nullptr);
    gemm_bf16<0><<<gD, blk, 0, stream>>>(lnbf, wb_q, nullptr, 0.f, qf, nullptr, nullptr, M, DMODEL, DMODEL);
    rope_transpose_k<<<M, blk, 0, stream>>>(qf, qb, 1);
    gemm_bf16<0><<<gD, blk, 0, stream>>>(lnbf, wb_k, nullptr, 0.f, qf, nullptr, nullptr, M, DMODEL, DMODEL);
    rope_transpose_k<<<M, blk, 0, stream>>>(qf, kb, 1);
    gemm_bf16<0><<<gD, blk, 0, stream>>>(lnbf, wb_v, nullptr, 0.f, qf, nullptr, nullptr, M, DMODEL, DMODEL);
    rope_transpose_k<<<M, blk, 0, stream>>>(qf, vb, 0);
    flash_attn_k<<<dim3(BATCH * NHEAD, TSEQ / 128), blk, 0, stream>>>(qb, kb, vb, attnbf);
    gemm_bf16<2><<<gD, blk, 0, stream>>>(attnbf, wb_o, nullptr, 1.f, nullptr, nullptr, xw, M, DMODEL, DMODEL);

    // === Conv module ===
    layernorm_k<<<M, blk, 0, stream>>>(xw, conv_lng, conv_lnb, lnbf, nullptr);
    gemm_bf16<0><<<gP, blk, 0, stream>>>(lnbf, wb_pw1, pw1_b, 0.f, ubuf, nullptr, nullptr, M, 2 * DMODEL, DMODEL);
    glu_k<<<4096, blk, 0, stream>>>(ubuf, hcbuf, M * DMODEL);
    dwconv_k<<<M, blk, 0, stream>>>(hcbuf, dw_w, dw_b, gbuf);
    bn_stats_k<<<DMODEL, blk, 0, stream>>>(gbuf, bn_mean, bn_inv);
    bn_apply_silu_k<<<4096, blk, 0, stream>>>(gbuf, bn_mean, bn_inv, bn_g, bn_b, sbf, M * DMODEL);
    gemm_bf16<2><<<gD, blk, 0, stream>>>(sbf, wb_pw2, pw2_b, 1.f, nullptr, nullptr, xw, M, DMODEL, DMODEL);

    // === FF2 (half-step) ===
    layernorm_k<<<M, blk, 0, stream>>>(xw, ff2_lng, ff2_lnb, lnbf, nullptr);
    gemm_bf16<1><<<gI, blk, 0, stream>>>(lnbf, wb_ff21, ff2_b1, 0.f, nullptr, inner_bf, nullptr, M, FFINNER, DMODEL);
    gemm_bf16<2><<<gD, blk, 0, stream>>>(inner_bf, wb_ff22, ff2_b2, 0.5f, nullptr, nullptr, xw, M, DMODEL, FFINNER);

    // === Output LayerNorm (f32) ===
    layernorm_k<<<M, blk, 0, stream>>>(xw, out_lng, out_lnb, nullptr, out);
}
